// NERBertCNNWithCRF_25735444038440
// MI455X (gfx1250) — matrix operand order repaired and checked by reading, not compiled, not run on hardware
//
#include <hip/hip_runtime.h>
#include <hip/hip_bf16.h>
#include <math.h>

typedef __bf16 bf16_t;
typedef __attribute__((ext_vector_type(16))) __bf16 v16bf;
typedef __attribute__((ext_vector_type(8)))  __bf16 v8bf;
typedef __attribute__((ext_vector_type(8)))  float  v8f;
typedef __attribute__((ext_vector_type(4)))  unsigned int v4u_t;
typedef __attribute__((ext_vector_type(8)))  int v8i_t;
typedef __attribute__((ext_vector_type(4)))  int v4i_t;

#define B_   32
#define S_   128
#define H_   768
#define L_   12
#define NH_  12
#define FF_  3072
#define T_   9
#define SP_  127
#define DH_  64
#define MTOK (B_*S_)   // 4096

// GEMM block tiling
#define BM 128
#define BN 128
#define BK 32
#define TILE_BYTES (BM * BK * 2)        // 8 KB per operand tile
#define BUF_BYTES  (2 * TILE_BYTES)     // A + B per buffer
#define SMEM_BYTES (2 * BUF_BYTES)      // double buffered = 32 KB

// ---------------------------------------------------------------------------
// WMMA fragment loaders (bf16, 16x16x32). k order used here (contiguous, k = 16*(L/16) + s, not ISA table order):
//  A (16x32): lane L holds row M=L%16; lanes<16 hold K 0..15,
//             lanes>=16 hold K 16..31.
//  B (32x16): lane L holds col N=L%16; lanes<16 hold K 0..15, lanes>=16 K 16..31.
// Operand tiles are stored [row/col][BK] so fragments are contiguous.
// ---------------------------------------------------------------------------
__device__ __forceinline__ v16bf load_a_frag(const bf16_t* rowp, int kk, int lane) {
  const int kbase = (lane >> 4) << 4;          // 0 or 16
  const bf16_t* p = rowp + kk + kbase;
  v8bf lo = *(const v8bf*)(p);
  v8bf hi = *(const v8bf*)(p + 8);
  v16bf f;
#pragma unroll
  for (int e = 0; e < 8; ++e) { f[e] = lo[e]; f[e + 8] = hi[e]; }
  return f;
}

__device__ __forceinline__ v16bf load_b_frag(const bf16_t* colp, int kk, int lane) {
  const int khalf = (lane >> 4) << 4;          // 0 or 16
  const bf16_t* p = colp + kk + khalf;
  v8bf lo = *(const v8bf*)(p);
  v8bf hi = *(const v8bf*)(p + 8);
  v16bf f;
#pragma unroll
  for (int e = 0; e < 8; ++e) { f[e] = lo[e]; f[e + 8] = hi[e]; }
  return f;
}

// ---------------------------------------------------------------------------
// TDM: async-load one [128 rows x 32 cols] bf16 tile (row stride K elements)
// into LDS at byte offset lds_off. D# per CDNA5 ISA ch.8 (group0 + group1,
// groups 2/3 zero => 2D tensor). data_size=1 (2B), tile_dim0=32, tile_dim1=128.
// ---------------------------------------------------------------------------
__device__ __forceinline__ void tdm_load_tile(const bf16_t* base, int rows_total,
                                              int K, int row_start, int kk,
                                              unsigned lds_off) {
  unsigned long long ga = (unsigned long long)(size_t)base +
      (((unsigned long long)(unsigned)row_start * (unsigned)K + (unsigned)kk) << 1);
  v4u_t g0;
  g0[0] = 1u;                                              // count=1 (valid), user mode
  g0[1] = lds_off;                                         // lds_addr (bytes)
  g0[2] = (unsigned)(ga & 0xFFFFFFFFull);                  // global_addr[31:0]
  g0[3] = (unsigned)((ga >> 32) & 0x1FFFFFFull) | (2u << 30); // addr[56:32] | type=2
  v8i_t g1;
  g1[0] = (int)(1u << 16);                                 // wg_mask=0, data_size=1 (2B)
  g1[1] = (int)(((unsigned)K & 0xFFFFu) << 16);            // tensor_dim0[15:0]
  g1[2] = (int)((((unsigned)K >> 16) & 0xFFFFu) |
                (((unsigned)rows_total & 0xFFFFu) << 16)); // td0[31:16] | td1[15:0]
  g1[3] = (int)((((unsigned)rows_total >> 16) & 0xFFFFu) |
                ((unsigned)BK << 16));                     // td1[31:16] | tile_dim0=32
  g1[4] = (int)(unsigned)BM;                               // tile_dim1=128, tile_dim2=0
  g1[5] = (int)(unsigned)K;                                // tensor_dim0_stride[31:0]
  g1[6] = 0;
  g1[7] = 0;
  v4i_t gz = {0, 0, 0, 0};
#if defined(__clang_major__) && (__clang_major__ >= 23)
  v8i_t gz8 = {0, 0, 0, 0, 0, 0, 0, 0};
  __builtin_amdgcn_tensor_load_to_lds(g0, g1, gz, gz, gz8, 0);
#else
  __builtin_amdgcn_tensor_load_to_lds(g0, g1, gz, gz, 0);
#endif
}

// ---------------------------------------------------------------------------
// Block-tiled GEMM: C[M,N] = A[M,K](bf16,row) @ Bt[N,K](bf16) + bias.
// Workgroup = 256 thr = 8 waves, block tile 128x128, K-step 32.
// Wave (2x4 grid) owns a 64x32 sub-tile = 4x2 accumulators -> 8 WMMA / K-step.
// A/B tiles staged in LDS via TDM, double buffered, s_wait_tensorcnt sync.
// epilogue: 0 = bias, 1 = bias + exact GELU. C (f32) / Cbf (bf16) optional.
// ---------------------------------------------------------------------------
__global__ __launch_bounds__(256)
void gemm_bf16_wmma_tiled(const bf16_t* __restrict__ A, const bf16_t* __restrict__ Bt,
                          const float* __restrict__ bias, float* __restrict__ C,
                          bf16_t* __restrict__ Cbf, int M, int N, int K, int epilogue) {
  extern __shared__ char smem[];           // [2] x { A: 8KB | B: 8KB }
  const int tid = threadIdx.x;
  const int wave = tid >> 5, lane = tid & 31;
  const int tilesN = N / BN;
  const int bm = blockIdx.x / tilesN, bn = blockIdx.x % tilesN;
  const int waveM = wave >> 2, waveN = wave & 3;   // 2 x 4 wave grid

  v8f acc[4][2] = {};

  if (wave == 0) {
    tdm_load_tile(A,  M, K, bm * BM, 0, 0u);
    tdm_load_tile(Bt, N, K, bn * BN, 0, (unsigned)TILE_BYTES);
  }

  int p = 0;
  for (int kk = 0; kk < K; kk += BK) {
    if (wave == 0) __builtin_amdgcn_s_wait_tensorcnt(0);
    __syncthreads();                       // tile p visible to all waves
    if (wave == 0 && (kk + BK) < K) {
      const unsigned nb = (unsigned)((p ^ 1) * BUF_BYTES);
      tdm_load_tile(A,  M, K, bm * BM, kk + BK, nb);
      tdm_load_tile(Bt, N, K, bn * BN, kk + BK, nb + (unsigned)TILE_BYTES);
    }
    const bf16_t* ldsA = (const bf16_t*)(smem + p * BUF_BYTES);
    const bf16_t* ldsB = (const bf16_t*)(smem + p * BUF_BYTES + TILE_BYTES);

    v16bf bfrag[2];
#pragma unroll
    for (int ni = 0; ni < 2; ++ni) {
      const int coll = waveN * 32 + ni * 16 + (lane & 15);
      bfrag[ni] = load_b_frag(ldsB + (size_t)coll * BK, 0, lane);
    }
#pragma unroll
    for (int mi = 0; mi < 4; ++mi) {
      const int rowl = waveM * 64 + mi * 16 + (lane & 15);
      v16bf afrag = load_a_frag(ldsA + (size_t)rowl * BK, 0, lane);
#pragma unroll
      for (int ni = 0; ni < 2; ++ni)
        acc[mi][ni] = __builtin_amdgcn_wmma_f32_16x16x32_bf16(
            false, afrag, false, bfrag[ni], (short)0, acc[mi][ni], false, false);
    }
    __syncthreads();                       // everyone done reading buffer p
    p ^= 1;
  }

  const int moff = (lane >> 4) << 3;       // 0 or 8
#pragma unroll
  for (int mi = 0; mi < 4; ++mi) {
#pragma unroll
    for (int ni = 0; ni < 2; ++ni) {
      const int n = bn * BN + waveN * 32 + ni * 16 + (lane & 15);
      const float bv = bias ? bias[n] : 0.0f;
#pragma unroll
      for (int r = 0; r < 8; ++r) {
        const int m = bm * BM + waveM * 64 + mi * 16 + moff + r;
        float v = acc[mi][ni][r] + bv;
        if (epilogue == 1) v = 0.5f * v * (1.0f + erff(v * 0.70710678118654752f));
        const size_t off = (size_t)m * N + n;
        if (C)   C[off]   = v;
        if (Cbf) Cbf[off] = (bf16_t)v;
      }
    }
  }
}

// ---------------------------------------------------------------------------
// Attention score: scores[b,h,q,k] = (Q.K^T)/8 + mask_bias ; WMMA over DH=64
// ---------------------------------------------------------------------------
__global__ __launch_bounds__(256)
void attn_score_kernel(const bf16_t* __restrict__ qkvbf, const int* __restrict__ amask,
                       float* __restrict__ scores) {
  const int wave = (blockIdx.x * blockDim.x + threadIdx.x) >> 5;
  const int lane = threadIdx.x & 31;
  const int wavesPerHead = (S_/16) * (S_/16);  // 64
  const int head = wave / wavesPerHead;
  if (head >= B_ * NH_) return;
  const int tile = wave % wavesPerHead;
  const int tq = tile >> 3, tk = tile & 7;
  const int b = head / NH_, h = head % NH_;

  const bf16_t* Qb = qkvbf + (size_t)(b * S_) * (3 * H_) + h * DH_;
  const bf16_t* Kb = Qb + H_;
  const int qrow = tq * 16 + (lane & 15);
  const int kcol = tk * 16 + (lane & 15);

  v8f acc = {};
#pragma unroll
  for (int kk = 0; kk < DH_; kk += 32) {
    v16bf a  = load_a_frag(Qb + (size_t)qrow * (3 * H_), kk, lane);
    v16bf bb = load_b_frag(Kb + (size_t)kcol * (3 * H_), kk, lane);
    acc = __builtin_amdgcn_wmma_f32_16x16x32_bf16(false, a, false, bb,
                                                  (short)0, acc, false, false);
  }
  const int moff = (lane >> 4) << 3;
  const float mb = (amask[b * S_ + kcol] > 0) ? 0.0f : -1000000000.0f;
#pragma unroll
  for (int r = 0; r < 8; ++r) {
    const int q = tq * 16 + moff + r;
    scores[(((size_t)(b * NH_ + h)) * S_ + q) * S_ + kcol] = acc[r] * 0.125f + mb;
  }
}

// ---------------------------------------------------------------------------
// Softmax along last dim (S=128), write bf16 probs
// ---------------------------------------------------------------------------
__global__ __launch_bounds__(128)
void softmax_kernel(const float* __restrict__ scores, bf16_t* __restrict__ probs) {
  const size_t row = blockIdx.x;
  const int t = threadIdx.x;
  const float v = scores[row * S_ + t];
  __shared__ float red[128];
  red[t] = v; __syncthreads();
  for (int o = 64; o > 0; o >>= 1) { if (t < o) red[t] = fmaxf(red[t], red[t + o]); __syncthreads(); }
  const float mx = red[0]; __syncthreads();
  const float e = __expf(v - mx);
  red[t] = e; __syncthreads();
  for (int o = 64; o > 0; o >>= 1) { if (t < o) red[t] += red[t + o]; __syncthreads(); }
  probs[row * S_ + t] = (bf16_t)(e / red[0]);
}

// ---------------------------------------------------------------------------
// P @ V : out_bf[b,q,h*64+d]; A = probs (row-major, stride S), B = V gathered
// ---------------------------------------------------------------------------
__global__ __launch_bounds__(256)
void attn_pv_kernel(const bf16_t* __restrict__ probs, const bf16_t* __restrict__ qkvbf,
                    bf16_t* __restrict__ obf) {
  const int wave = (blockIdx.x * blockDim.x + threadIdx.x) >> 5;
  const int lane = threadIdx.x & 31;
  const int wavesPerHead = (S_/16) * (DH_/16); // 32
  const int head = wave / wavesPerHead;
  if (head >= B_ * NH_) return;
  const int tile = wave % wavesPerHead;
  const int tq = tile >> 2, td = tile & 3;
  const int b = head / NH_, h = head % NH_;

  const bf16_t* P = probs + (size_t)head * S_ * S_;
  const int qrow = tq * 16 + (lane & 15);
  const int dcol = td * 16 + (lane & 15);
  const int khalf = (lane >> 4) << 4;

  v8f acc = {};
#pragma unroll
  for (int kk = 0; kk < S_; kk += 32) {
    v16bf a = load_a_frag(P + (size_t)qrow * S_, kk, lane);
    v16bf bb;
#pragma unroll
    for (int e = 0; e < 16; ++e) {
      const int sk = kk + khalf + e;
      bb[e] = qkvbf[((size_t)(b * S_ + sk)) * (3 * H_) + 2 * H_ + h * DH_ + dcol];
    }
    acc = __builtin_amdgcn_wmma_f32_16x16x32_bf16(false, a, false, bb,
                                                  (short)0, acc, false, false);
  }
  const int moff = (lane >> 4) << 3;
#pragma unroll
  for (int r = 0; r < 8; ++r) {
    const int q = tq * 16 + moff + r;
    obf[((size_t)(b * S_ + q)) * H_ + h * DH_ + dcol] = (bf16_t)acc[r];
  }
}

// ---------------------------------------------------------------------------
// Embedding gather + LayerNorm (one block per token, 256 thr, 3 elems/thr)
// ---------------------------------------------------------------------------
__global__ __launch_bounds__(256)
void embed_ln_kernel(const int* __restrict__ ids, const int* __restrict__ tt,
                     const float* __restrict__ tok, const float* __restrict__ pos,
                     const float* __restrict__ typ, const float* __restrict__ g,
                     const float* __restrict__ be, float* __restrict__ xo,
                     bf16_t* __restrict__ xbf) {
  const int token = blockIdx.x;
  const int s = token & (S_ - 1);
  const int id = ids[token], t2 = tt[token];
  float v[3]; float lsum = 0.f, lsq = 0.f;
#pragma unroll
  for (int j = 0; j < 3; ++j) {
    const int i = threadIdx.x + j * 256;
    const float val = tok[(size_t)id * H_ + i] + pos[s * H_ + i] + typ[t2 * H_ + i];
    v[j] = val; lsum += val; lsq += val * val;
  }
  __shared__ float rs[256], rq[256];
  rs[threadIdx.x] = lsum; rq[threadIdx.x] = lsq; __syncthreads();
  for (int o = 128; o > 0; o >>= 1) {
    if (threadIdx.x < o) { rs[threadIdx.x] += rs[threadIdx.x + o]; rq[threadIdx.x] += rq[threadIdx.x + o]; }
    __syncthreads();
  }
  const float mean = rs[0] * (1.0f / H_);
  const float var  = rq[0] * (1.0f / H_) - mean * mean;
  const float inv  = rsqrtf(var + 1e-12f);
#pragma unroll
  for (int j = 0; j < 3; ++j) {
    const int i = threadIdx.x + j * 256;
    const float o_ = (v[j] - mean) * inv * g[i] + be[i];
    xo[(size_t)token * H_ + i] = o_;
    xbf[(size_t)token * H_ + i] = (bf16_t)o_;
  }
}

// ---------------------------------------------------------------------------
// Residual add + LayerNorm (in-place safe: reads precede writes)
// ---------------------------------------------------------------------------
__global__ __launch_bounds__(256)
void residual_ln_kernel(const float* __restrict__ resid, const float* __restrict__ y,
                        const float* __restrict__ g, const float* __restrict__ be,
                        float* __restrict__ xo, bf16_t* __restrict__ xbf) {
  const int token = blockIdx.x;
  float v[3]; float lsum = 0.f, lsq = 0.f;
#pragma unroll
  for (int j = 0; j < 3; ++j) {
    const int i = threadIdx.x + j * 256;
    const float val = resid[(size_t)token * H_ + i] + y[(size_t)token * H_ + i];
    v[j] = val; lsum += val; lsq += val * val;
  }
  __shared__ float rs[256], rq[256];
  rs[threadIdx.x] = lsum; rq[threadIdx.x] = lsq; __syncthreads();
  for (int o = 128; o > 0; o >>= 1) {
    if (threadIdx.x < o) { rs[threadIdx.x] += rs[threadIdx.x + o]; rq[threadIdx.x] += rq[threadIdx.x + o]; }
    __syncthreads();
  }
  const float mean = rs[0] * (1.0f / H_);
  const float var  = rq[0] * (1.0f / H_) - mean * mean;
  const float inv  = rsqrtf(var + 1e-12f);
#pragma unroll
  for (int j = 0; j < 3; ++j) {
    const int i = threadIdx.x + j * 256;
    const float o_ = (v[j] - mean) * inv * g[i] + be[i];
    xo[(size_t)token * H_ + i] = o_;
    xbf[(size_t)token * H_ + i] = (bf16_t)o_;
  }
}

// ---------------------------------------------------------------------------
// Weight convert fp32[K,N] -> bf16[N,K] (transposed for B-fragment layout)
// ---------------------------------------------------------------------------
__global__ __launch_bounds__(256)
void wt_transpose_kernel(const float* __restrict__ W, bf16_t* __restrict__ Bt,
                         int K, int N) {
  const int idx = blockIdx.x * blockDim.x + threadIdx.x;
  if (idx >= K * N) return;
  const int n = idx % N, k = idx / N;
  Bt[(size_t)n * K + k] = (bf16_t)W[(size_t)k * N + n];
}

// ---------------------------------------------------------------------------
// Conv1: NCH, stride 2, k=5.  hs = x[:,1:,:]  -> out (B,127,382)
// ---------------------------------------------------------------------------
__global__ __launch_bounds__(256)
void conv1_kernel(const float* __restrict__ x, const float* __restrict__ w,
                  const float* __restrict__ bias, float* __restrict__ out) {
  const int idx = blockIdx.x * blockDim.x + threadIdx.x;
  const int TOT = B_ * SP_ * 382;
  if (idx >= TOT) return;
  const int t = idx % 382, o = (idx / 382) % SP_, b = idx / (382 * SP_);
  float sum = bias[o];
  const float* xb = x + ((size_t)(b * S_ + 1)) * H_;
  const float* wo = w + (size_t)o * SP_ * 5;
  for (int i = 0; i < SP_; ++i) {
    const float* xr = xb + (size_t)i * H_ + 2 * t;
    const float* wr = wo + i * 5;
#pragma unroll
    for (int k = 0; k < 5; ++k) sum += xr[k] * wr[k];
  }
  out[idx] = sum;
}

// Conv2: stride 2, k=3, dilation 2 -> out (B,127,189)
__global__ __launch_bounds__(256)
void conv2_kernel(const float* __restrict__ c1, const float* __restrict__ w,
                  const float* __restrict__ bias, float* __restrict__ out) {
  const int idx = blockIdx.x * blockDim.x + threadIdx.x;
  const int TOT = B_ * SP_ * 189;
  if (idx >= TOT) return;
  const int t = idx % 189, o = (idx / 189) % SP_, b = idx / (189 * SP_);
  float sum = bias[o];
  const float* cb = c1 + (size_t)(b * SP_) * 382;
  const float* wo = w + (size_t)o * SP_ * 3;
  for (int i = 0; i < SP_; ++i) {
    const float* cr = cb + (size_t)i * 382 + 2 * t;
#pragma unroll
    for (int k = 0; k < 3; ++k) sum += cr[2 * k] * wo[i * 3 + k];
  }
  out[idx] = sum;
}

// logits[b,p,tag] = sum_j c2[b,p,j]*lin_w[j,tag] + lin_b[tag]
__global__ __launch_bounds__(256)
void linear_kernel(const float* __restrict__ c2, const float* __restrict__ lw,
                   const float* __restrict__ lb, float* __restrict__ logits) {
  const int idx = blockIdx.x * blockDim.x + threadIdx.x;
  const int TOT = B_ * SP_ * T_;
  if (idx >= TOT) return;
  const int tag = idx % T_, p = (idx / T_) % SP_, b = idx / (T_ * SP_);
  float sum = lb[tag];
  const float* cr = c2 + ((size_t)(b * SP_ + p)) * 189;
  for (int j = 0; j < 189; ++j) sum += cr[j] * lw[j * T_ + tag];
  logits[idx] = sum;
}

// ---------------------------------------------------------------------------
// CRF forward (logsumexp) -> norm[b]
// ---------------------------------------------------------------------------
__global__ void crf_norm_kernel(const float* __restrict__ logits, const int* __restrict__ amask,
                                const float* __restrict__ start, const float* __restrict__ endv,
                                const float* __restrict__ trans, float* __restrict__ norm_out) {
  const int b = blockIdx.x, j = threadIdx.x;
  __shared__ float s[T_];
  if (j < T_) s[j] = start[j] + logits[((size_t)b * SP_) * T_ + j];
  __syncthreads();
  for (int t = 1; t < SP_; ++t) {
    float ns = 0.f;
    if (j < T_) {
      float mx = -1e30f;
      for (int i = 0; i < T_; ++i) mx = fmaxf(mx, s[i] + trans[i * T_ + j]);
      float sum = 0.f;
      for (int i = 0; i < T_; ++i) sum += __expf(s[i] + trans[i * T_ + j] - mx);
      ns = mx + __logf(sum) + logits[((size_t)b * SP_ + t) * T_ + j];
    }
    const int m = amask[b * S_ + t + 1];
    __syncthreads();
    if (j < T_ && m) s[j] = ns;
    __syncthreads();
  }
  if (j == 0) {
    float mx = -1e30f;
    for (int i = 0; i < T_; ++i) mx = fmaxf(mx, s[i] + endv[i]);
    float sum = 0.f;
    for (int i = 0; i < T_; ++i) sum += __expf(s[i] + endv[i] - mx);
    norm_out[b] = mx + __logf(sum);
  }
}

// CRF gold path score -> score[b]   (1 block, thread b per batch)
__global__ void crf_score_kernel(const float* __restrict__ logits, const int* __restrict__ tags,
                                 const int* __restrict__ amask, const float* __restrict__ start,
                                 const float* __restrict__ endv, const float* __restrict__ trans,
                                 float* __restrict__ score_out) {
  const int b = threadIdx.x;
  if (b >= B_) return;
  const int t0 = tags[b * SP_];
  float sc = start[t0] + logits[((size_t)b * SP_) * T_ + t0];
  int cnt = 0;
  for (int t = 0; t < SP_; ++t) cnt += (amask[b * S_ + 1 + t] != 0);
  for (int t = 1; t < SP_; ++t) {
    const float m = (amask[b * S_ + 1 + t] != 0) ? 1.0f : 0.0f;
    const int pa = tags[b * SP_ + t - 1], cu = tags[b * SP_ + t];
    sc += m * (trans[pa * T_ + cu] + logits[((size_t)b * SP_ + t) * T_ + cu]);
  }
  const int lt = tags[b * SP_ + (cnt - 1)];
  score_out[b] = sc + endv[lt];
}

// Viterbi: forward + LDS history + backtrack; path as floats into out
__global__ void viterbi_kernel(const float* __restrict__ logits, const int* __restrict__ amask,
                               const float* __restrict__ start, const float* __restrict__ endv,
                               const float* __restrict__ trans, float* __restrict__ path_out) {
  const int b = blockIdx.x, j = threadIdx.x;
  __shared__ float s[T_];
  __shared__ short hist[(SP_ - 1) * T_];
  if (j < T_) s[j] = start[j] + logits[((size_t)b * SP_) * T_ + j];
  __syncthreads();
  for (int t = 1; t < SP_; ++t) {
    float ns = 0.f; int bi = j;
    if (j < T_) {
      float mx = -1e30f;
      for (int i = 0; i < T_; ++i) {
        const float c = s[i] + trans[i * T_ + j];
        if (c > mx) { mx = c; bi = i; }
      }
      ns = mx + logits[((size_t)b * SP_ + t) * T_ + j];
    }
    const int m = amask[b * S_ + t + 1];
    __syncthreads();
    if (j < T_) {
      if (m) { s[j] = ns; hist[(t - 1) * T_ + j] = (short)bi; }
      else   { hist[(t - 1) * T_ + j] = (short)j; }
    }
    __syncthreads();
  }
  if (j == 0) {
    float mx = -1e30f; int tag = 0;
    for (int i = 0; i < T_; ++i) {
      const float c = s[i] + endv[i];
      if (c > mx) { mx = c; tag = i; }
    }
    path_out[b * SP_ + SP_ - 1] = (float)tag;
    for (int t = SP_ - 2; t >= 0; --t) {
      tag = hist[t * T_ + tag];
      path_out[b * SP_ + t] = (float)tag;
    }
  }
}

// llh = mean(score - norm) -> out[0]
__global__ void llh_kernel(const float* __restrict__ score, const float* __restrict__ norm,
                           float* __restrict__ out) {
  const int t = threadIdx.x;
  __shared__ float r[32];
  r[t] = (t < B_) ? (score[t] - norm[t]) : 0.f;
  __syncthreads();
  for (int o = 16; o > 0; o >>= 1) { if (t < o) r[t] += r[t + o]; __syncthreads(); }
  if (t == 0) out[0] = r[0] * (1.0f / B_);
}

// ---------------------------------------------------------------------------
static inline int cdiv(long long a, long long b) { return (int)((a + b - 1) / b); }

extern "C" void kernel_launch(void* const* d_in, const int* in_sizes, int n_in,
                              void* d_out, int out_size, void* d_ws, size_t ws_size,
                              hipStream_t stream) {
  (void)in_sizes; (void)n_in; (void)out_size; (void)ws_size;
  const int*   input_ids = (const int*)  d_in[0];
  const int*   amask     = (const int*)  d_in[1];
  const int*   ttids     = (const int*)  d_in[2];
  const int*   y_true    = (const int*)  d_in[3];
  const float* tok_emb   = (const float*)d_in[4];
  const float* pos_emb   = (const float*)d_in[5];
  const float* type_emb  = (const float*)d_in[6];
  const float* emb_g     = (const float*)d_in[7];
  const float* emb_b     = (const float*)d_in[8];
  const float* Wqkv      = (const float*)d_in[9];
  const float* bqkv      = (const float*)d_in[10];
  const float* Wo        = (const float*)d_in[11];
  const float* bo        = (const float*)d_in[12];
  const float* ln1g      = (const float*)d_in[13];
  const float* ln1b      = (const float*)d_in[14];
  const float* Wff1      = (const float*)d_in[15];
  const float* bff1      = (const float*)d_in[16];
  const float* Wff2      = (const float*)d_in[17];
  const float* bff2      = (const float*)d_in[18];
  const float* ln2g      = (const float*)d_in[19];
  const float* ln2b      = (const float*)d_in[20];
  const float* conv1w    = (const float*)d_in[21];
  const float* conv1b    = (const float*)d_in[22];
  const float* conv2w    = (const float*)d_in[23];
  const float* conv2b    = (const float*)d_in[24];
  const float* lin_w     = (const float*)d_in[25];
  const float* lin_b     = (const float*)d_in[26];
  const float* crf_start = (const float*)d_in[27];
  const float* crf_end   = (const float*)d_in[28];
  const float* crf_trans = (const float*)d_in[29];
  float* out = (float*)d_out;

  // workspace carve (256B aligned)
  char* base = (char*)d_ws;
  size_t off = 0;
  auto carve = [&](size_t bytes) -> char* {
    char* p = base + off;
    off = (off + bytes + 255) & ~(size_t)255;
    return p;
  };
  float*  x       = (float*) carve((size_t)MTOK * H_ * 4);
  bf16_t* xbf     = (bf16_t*)carve((size_t)MTOK * H_ * 2);
  bf16_t* qkvbf   = (bf16_t*)carve((size_t)MTOK * 3 * H_ * 2);
  float*  scores  = (float*) carve((size_t)B_ * NH_ * S_ * S_ * 4);
  bf16_t* probsbf = (bf16_t*)carve((size_t)B_ * NH_ * S_ * S_ * 2);
  bf16_t* obf     = (bf16_t*)carve((size_t)MTOK * H_ * 2);
  float*  tmp     = (float*) carve((size_t)MTOK * H_ * 4);
  bf16_t* ffbf    = (bf16_t*)carve((size_t)MTOK * FF_ * 2);
  bf16_t* WqkvT   = (bf16_t*)carve((size_t)3 * H_ * H_ * 2);
  bf16_t* WoT     = (bf16_t*)carve((size_t)H_ * H_ * 2);
  bf16_t* Wff1T   = (bf16_t*)carve((size_t)FF_ * H_ * 2);
  bf16_t* Wff2T   = (bf16_t*)carve((size_t)H_ * FF_ * 2);
  float*  c1buf   = (float*) carve((size_t)B_ * SP_ * 382 * 4);
  float*  c2buf   = (float*) carve((size_t)B_ * SP_ * 189 * 4);
  float*  logits  = (float*) carve((size_t)B_ * SP_ * T_ * 4);
  float*  normb   = (float*) carve((size_t)B_ * 4);
  float*  scoreb  = (float*) carve((size_t)B_ * 4);

  // embeddings + LN
  embed_ln_kernel<<<MTOK, 256, 0, stream>>>(input_ids, ttids, tok_emb, pos_emb,
                                            type_emb, emb_g, emb_b, x, xbf);

  auto gemm_grid = [&](int M, int N) { return (M / BM) * (N / BN); };

  for (int l = 0; l < L_; ++l) {
    const float* Wqkv_l = Wqkv + (size_t)l * H_ * 3 * H_;
    const float* bqkv_l = bqkv + (size_t)l * 3 * H_;
    const float* Wo_l   = Wo   + (size_t)l * H_ * H_;
    const float* bo_l   = bo   + (size_t)l * H_;
    const float* g1     = ln1g + (size_t)l * H_;
    const float* b1     = ln1b + (size_t)l * H_;
    const float* W1_l   = Wff1 + (size_t)l * H_ * FF_;
    const float* c1_l   = bff1 + (size_t)l * FF_;
    const float* W2_l   = Wff2 + (size_t)l * FF_ * H_;
    const float* c2_l   = bff2 + (size_t)l * H_;
    const float* g2     = ln2g + (size_t)l * H_;
    const float* b2     = ln2b + (size_t)l * H_;

    // QKV projection -> qkvbf
    wt_transpose_kernel<<<cdiv((long long)H_ * 3 * H_, 256), 256, 0, stream>>>(Wqkv_l, WqkvT, H_, 3 * H_);
    gemm_bf16_wmma_tiled<<<gemm_grid(MTOK, 3 * H_), 256, SMEM_BYTES, stream>>>(
        xbf, WqkvT, bqkv_l, nullptr, qkvbf, MTOK, 3 * H_, H_, 0);

    // attention
    attn_score_kernel<<<cdiv((long long)B_ * NH_ * 64 * 32, 256), 256, 0, stream>>>(qkvbf, amask, scores);
    softmax_kernel<<<B_ * NH_ * S_, 128, 0, stream>>>(scores, probsbf);
    attn_pv_kernel<<<cdiv((long long)B_ * NH_ * 32 * 32, 256), 256, 0, stream>>>(probsbf, qkvbf, obf);

    // output projection + residual LN1
    wt_transpose_kernel<<<cdiv((long long)H_ * H_, 256), 256, 0, stream>>>(Wo_l, WoT, H_, H_);
    gemm_bf16_wmma_tiled<<<gemm_grid(MTOK, H_), 256, SMEM_BYTES, stream>>>(
        obf, WoT, bo_l, tmp, nullptr, MTOK, H_, H_, 0);
    residual_ln_kernel<<<MTOK, 256, 0, stream>>>(x, tmp, g1, b1, x, xbf);

    // FFN
    wt_transpose_kernel<<<cdiv((long long)H_ * FF_, 256), 256, 0, stream>>>(W1_l, Wff1T, H_, FF_);
    gemm_bf16_wmma_tiled<<<gemm_grid(MTOK, FF_), 256, SMEM_BYTES, stream>>>(
        xbf, Wff1T, c1_l, nullptr, ffbf, MTOK, FF_, H_, 1);
    wt_transpose_kernel<<<cdiv((long long)FF_ * H_, 256), 256, 0, stream>>>(W2_l, Wff2T, FF_, H_);
    gemm_bf16_wmma_tiled<<<gemm_grid(MTOK, H_), 256, SMEM_BYTES, stream>>>(
        ffbf, Wff2T, c2_l, tmp, nullptr, MTOK, H_, FF_, 0);
    residual_ln_kernel<<<MTOK, 256, 0, stream>>>(x, tmp, g2, b2, x, xbf);
  }

  // conv head
  conv1_kernel<<<cdiv((long long)B_ * SP_ * 382, 256), 256, 0, stream>>>(x, conv1w, conv1b, c1buf);
  conv2_kernel<<<cdiv((long long)B_ * SP_ * 189, 256), 256, 0, stream>>>(c1buf, conv2w, conv2b, c2buf);
  linear_kernel<<<cdiv((long long)B_ * SP_ * T_, 256), 256, 0, stream>>>(c2buf, lin_w, lin_b, logits);

  // CRF
  crf_norm_kernel<<<B_, 32, 0, stream>>>(logits, amask, crf_start, crf_end, crf_trans, normb);
  crf_score_kernel<<<1, 32, 0, stream>>>(logits, y_true, amask, crf_start, crf_end, crf_trans, scoreb);
  viterbi_kernel<<<B_, 32, 0, stream>>>(logits, amask, crf_start, crf_end, crf_trans, out + 1);
  llh_kernel<<<1, 32, 0, stream>>>(scoreb, normb, out);
}
